// SupervisedMoEPredictor_19963007992299
// MI455X (gfx1250) — matrix operand order repaired and checked by reading, not compiled, not run on hardware
//
#include <hip/hip_runtime.h>
#include <math.h>

typedef __attribute__((ext_vector_type(16))) _Float16 v16h;
typedef __attribute__((ext_vector_type(8)))  _Float16 v8h;
typedef __attribute__((ext_vector_type(8)))  float    v8f;

#define NTOK   49152      // B*M tokens
#define EXP    5
#define TC     120        // T*2
#define ROWS   48         // token rows per block
#define NTHR   96         // 3 waves of 32
#define LXP    136        // 128+8 halves pitch (bank-conflict pad)
#define HP     264        // 256+8 halves pitch

// ---------------- device helpers ----------------

// Branch-free GELU (exact erf form, Abramowitz–Stegun 7.1.26, |err| < 1.5e-7).
// One v_rcp, one v_exp, straight-line FMAs: no EXEC-mask divergence in the
// WMMA hot loop (erff() lowers to saveexec branches which serialized round 1).
__device__ __forceinline__ float gelu_exact(float v) {
  const float z = fabsf(v) * 0.70710678118654752f;
  const float t = __builtin_amdgcn_rcpf(1.0f + 0.3275911f * z);
  const float poly = t * (0.254829592f + t * (-0.284496736f + t * (1.421413741f +
                     t * (-1.453152027f + t * 1.061405429f))));
  float erfz = 1.0f - poly * __expf(-z * z);
  erfz = copysignf(erfz, v);
  return 0.5f * v * (1.0f + erfz);
}

// One 16x16 C tile: A rows from LDS (arow = this lane's row base),
// B cols from global f16 weights transposed to [Nout][K].
__device__ __forceinline__ v8f wmma_rowcol(const _Float16* arow,
                                           const _Float16* __restrict__ bcol,
                                           int K, int hg) {
  v8f acc = {};
  for (int k0 = 0; k0 < K; k0 += 32) {
    // A 16x32 f16 layout: lane half-group hg holds 16 consecutive K = k0+hg*16 .. +15
    v8h alo = *(const v8h*)(arow + k0 + hg * 16);
    v8h ahi = *(const v8h*)(arow + k0 + hg * 16 + 8);
    union { v16h v; v8h h[2]; } ua;
    ua.h[0] = alo; ua.h[1] = ahi;
    // B 32x16 f16 layout: lane half-group hg holds 16 consecutive K values
    v16h b = *(const v16h*)(bcol + k0 + hg * 16);
    acc = __builtin_amdgcn_wmma_f32_16x16x32_f16(false, ua.v, false, b,
                                                 (short)0, acc, false, false);
  }
  return acc;
}

// ---------------- weight convert: f32 [K][Nn] -> f16 [Np][K] (transposed, zero-padded)

__global__ __launch_bounds__(256) void convert_w(const float* __restrict__ src,
                                                 _Float16* __restrict__ dst,
                                                 int K, int Nn, int Np) {
  int idx = blockIdx.x * 256 + threadIdx.x;
  if (idx >= Np * K) return;
  int n = idx / K, k = idx - n * K;
  float v = (n < Nn) ? src[(size_t)k * Nn + n] : 0.0f;
  dst[idx] = (_Float16)v;
}

// ---------------- generic 3-layer MLP (gelu, gelu, linear) via WMMA ----------------

__global__ __launch_bounds__(NTHR) void mlp3_wmma(
    const float* __restrict__ x,
    const _Float16* __restrict__ w1t, const float* __restrict__ b1, int H1,
    const _Float16* __restrict__ w2t, const float* __restrict__ b2, int H2,
    const _Float16* __restrict__ w3t, const float* __restrict__ b3, int H3p, int H3r,
    float* __restrict__ out, int ostride)
{
  __shared__ _Float16 lx[ROWS][LXP];
  __shared__ _Float16 h1s[ROWS][HP];
  __shared__ _Float16 h2s[ROWS][HP];
  const int tid  = threadIdx.x;
  const int wave = tid >> 5, lane = tid & 31;
  const int hg   = lane >> 4, l16 = lane & 15;
  const int mrow = wave * 16;
  const size_t row0 = (size_t)blockIdx.x * ROWS;

  // stage x tile f32 -> f16 LDS (N divisible by 48, no guards needed)
  for (int i = tid; i < ROWS * 32; i += NTHR) {
    int r = i >> 5, cc = (i & 31) << 2;
    const float4 v = *(const float4*)(x + (row0 + r) * 128 + cc);
    lx[r][cc]     = (_Float16)v.x;
    lx[r][cc + 1] = (_Float16)v.y;
    lx[r][cc + 2] = (_Float16)v.z;
    lx[r][cc + 3] = (_Float16)v.w;
  }
  __syncthreads();

  // ---- layer 1: [48x128] @ [128xH1] -> gelu -> h1s
  const _Float16* a1 = &lx[mrow + l16][0];
  for (int n0 = 0; n0 < H1; n0 += 16) {
    v8f acc = wmma_rowcol(a1, w1t + (size_t)(n0 + l16) * 128, 128, hg);
    const float bias = b1[n0 + l16];
#pragma unroll
    for (int r = 0; r < 8; ++r)
      h1s[mrow + r + 8 * hg][n0 + l16] = (_Float16)gelu_exact(acc[r] + bias);
  }
  __syncthreads();

  // ---- layer 2: [48xH1] @ [H1xH2] -> gelu -> h2s
  const _Float16* a2 = &h1s[mrow + l16][0];
  for (int n0 = 0; n0 < H2; n0 += 16) {
    v8f acc = wmma_rowcol(a2, w2t + (size_t)(n0 + l16) * H1, H1, hg);
    const float bias = b2[n0 + l16];
#pragma unroll
    for (int r = 0; r < 8; ++r)
      h2s[mrow + r + 8 * hg][n0 + l16] = (_Float16)gelu_exact(acc[r] + bias);
  }
  __syncthreads();

  // ---- layer 3: [48xH2] @ [H2xH3] + bias -> global (only first H3r cols real)
  const _Float16* a3 = &h2s[mrow + l16][0];
  for (int n0 = 0; n0 < H3p; n0 += 16) {
    v8f acc = wmma_rowcol(a3, w3t + (size_t)(n0 + l16) * H2, H2, hg);
    int col = n0 + l16;
    if (col < H3r) {
      const float bias = b3[col];
#pragma unroll
      for (int r = 0; r < 8; ++r)
        out[(row0 + mrow + r + 8 * hg) * (size_t)ostride + col] = acc[r] + bias;
    }
  }
}

// ---------------- router post: softmax, partial sums, top-2 ----------------

__global__ __launch_bounds__(256) void router_post(
    const float* __restrict__ logits, float* __restrict__ tkp, int* __restrict__ tki,
    float* __restrict__ tkidx_f, float* __restrict__ partials)
{
  const int tid = threadIdx.x;
  const int t = blockIdx.x * 256 + tid;
  float lg[EXP];
#pragma unroll
  for (int e = 0; e < EXP; ++e) lg[e] = logits[(size_t)t * EXP + e];
  float mx = lg[0];
#pragma unroll
  for (int e = 1; e < EXP; ++e) mx = fmaxf(mx, lg[e]);
  float p[EXP]; float s = 0.f;
#pragma unroll
  for (int e = 0; e < EXP; ++e) { p[e] = expf(lg[e] - mx); s += p[e]; }
  const float inv = 1.f / s;
#pragma unroll
  for (int e = 0; e < EXP; ++e) p[e] *= inv;

  // deterministic block reduction of probs for aux loss
  __shared__ float red[256];
  for (int e = 0; e < EXP; ++e) {
    red[tid] = p[e];
    __syncthreads();
    for (int st = 128; st > 0; st >>= 1) {
      if (tid < st) red[tid] += red[tid + st];
      __syncthreads();
    }
    if (tid == 0) partials[(size_t)blockIdx.x * EXP + e] = red[0];
    __syncthreads();
  }

  // top-2 (ties -> lowest index, matches stable top_k)
  int i0 = 0;
#pragma unroll
  for (int e = 1; e < EXP; ++e) if (lg[e] > lg[i0]) i0 = e;
  int i1 = (i0 == 0) ? 1 : 0;
#pragma unroll
  for (int e = 0; e < EXP; ++e) if (e != i0 && lg[e] > lg[i1]) i1 = e;
  const float e0 = expf(lg[i0] - lg[i0]), e1 = expf(lg[i1] - lg[i0]);
  const float is2 = 1.f / (e0 + e1);
  tkp[t * 2] = e0 * is2;  tkp[t * 2 + 1] = e1 * is2;
  tki[t * 2] = i0;        tki[t * 2 + 1] = i1;
  tkidx_f[t * 2] = (float)i0;  tkidx_f[t * 2 + 1] = (float)i1;
}

__global__ void aux_kernel(const float* __restrict__ partials, int nblk,
                           float* __restrict__ aux)
{
  if (threadIdx.x == 0 && blockIdx.x == 0) {
    float a[EXP] = {};
    for (int b = 0; b < nblk; ++b)
      for (int e = 0; e < EXP; ++e) a[e] += partials[(size_t)b * EXP + e];
    float ent = 0.f, l2 = 0.f;
    for (int e = 0; e < EXP; ++e) {
      float ap = a[e] / (float)NTOK;
      ent -= ap * logf(ap + 1e-8f);
      float d = ap - 1.f / (float)EXP;
      l2 += d * d;
    }
    l2 /= (float)EXP;
    aux[0] = -ent * 0.01f + 0.01f * l2;
  }
}

// ---------------- top-k combine: one wave per token ----------------

__global__ __launch_bounds__(256) void combine_kernel(
    const int* __restrict__ tki, const float* __restrict__ tkp,
    const float* __restrict__ sh_traj, const float* __restrict__ sh_score,
    const float* __restrict__ traj_all, const float* __restrict__ score_all,
    float* __restrict__ ftraj, float* __restrict__ fscore)
{
  const int token = blockIdx.x * 8 + (threadIdx.x >> 5);
  const int lane = threadIdx.x & 31;
  const int i0 = tki[token * 2], i1 = tki[token * 2 + 1];
  const float p0 = tkp[token * 2], p1 = tkp[token * 2 + 1];
  if (lane == 0) {
    float us = p0 * score_all[(size_t)i0 * NTOK + token] +
               p1 * score_all[(size_t)i1 * NTOK + token];
    fscore[token] = 0.3f * sh_score[token] + 0.7f * us;
  }
  const size_t g0 = ((size_t)i0 * NTOK + token) * TC;
  const size_t g1 = ((size_t)i1 * NTOK + token) * TC;
  const size_t st = (size_t)token * TC;
  for (int c = lane; c < TC; c += 32) {
    float u = p0 * traj_all[g0 + c] + p1 * traj_all[g1 + c];
    ftraj[st + c] = 0.3f * sh_traj[st + c] + 0.7f * u;
  }
}

// ---------------- host ----------------

extern "C" void kernel_launch(void* const* d_in, const int* in_sizes, int n_in,
                              void* d_out, int out_size, void* d_ws, size_t ws_size,
                              hipStream_t stream)
{
  (void)in_sizes; (void)n_in; (void)out_size; (void)ws_size;
  const float* x    = (const float*)d_in[0];
  const float* rw1  = (const float*)d_in[1];  const float* rb1 = (const float*)d_in[2];
  const float* rw2  = (const float*)d_in[3];  const float* rb2 = (const float*)d_in[4];
  const float* rw3  = (const float*)d_in[5];  const float* rb3 = (const float*)d_in[6];
  const float* stw1 = (const float*)d_in[7];  const float* stb1 = (const float*)d_in[8];
  const float* stw2 = (const float*)d_in[9];  const float* stb2 = (const float*)d_in[10];
  const float* stw3 = (const float*)d_in[11]; const float* stb3 = (const float*)d_in[12];
  const float* ssw1 = (const float*)d_in[13]; const float* ssb1 = (const float*)d_in[14];
  const float* ssw2 = (const float*)d_in[15]; const float* ssb2 = (const float*)d_in[16];
  const float* ssw3 = (const float*)d_in[17]; const float* ssb3 = (const float*)d_in[18];
  const float* etw1 = (const float*)d_in[19]; const float* etb1 = (const float*)d_in[20];
  const float* etw2 = (const float*)d_in[21]; const float* etb2 = (const float*)d_in[22];
  const float* etw3 = (const float*)d_in[23]; const float* etb3 = (const float*)d_in[24];
  const float* esw1 = (const float*)d_in[25]; const float* esb1 = (const float*)d_in[26];
  const float* esw2 = (const float*)d_in[27]; const float* esb2 = (const float*)d_in[28];
  const float* esw3 = (const float*)d_in[29]; const float* esb3 = (const float*)d_in[30];

  const size_t N = NTOK;
  float* out        = (float*)d_out;
  float* o_ftraj    = out;
  float* o_fscore   = o_ftraj + N * TC;
  float* o_logits   = o_fscore + N;
  float* o_topk     = o_logits + N * EXP;
  float* o_aux      = o_topk + N * 2;
  float* o_traj_all = o_aux + 1;
  float* o_score_all= o_traj_all + (size_t)EXP * N * TC;

  // workspace carve-up
  char* wsb = (char*)d_ws;
  size_t off = 0;
  auto allocB = [&](size_t bytes) -> void* {
    off = (off + 255) & ~(size_t)255;
    void* p = wsb + off; off += bytes; return p;
  };
  auto allocH = [&](size_t n) { return (_Float16*)allocB(n * sizeof(_Float16)); };
  auto allocF = [&](size_t n) { return (float*)allocB(n * sizeof(float)); };

  _Float16* r_w1t  = allocH(256 * 128);
  _Float16* r_w2t  = allocH(128 * 256);
  _Float16* r_w3t  = allocH(16 * 128);
  _Float16* st_w1t = allocH(256 * 128);
  _Float16* st_w2t = allocH(256 * 256);
  _Float16* st_w3t = allocH(128 * 256);
  _Float16* ss_w1t = allocH(128 * 128);
  _Float16* ss_w2t = allocH(64 * 128);
  _Float16* ss_w3t = allocH(16 * 64);
  _Float16 *et_w1t[EXP], *et_w2t[EXP], *et_w3t[EXP];
  _Float16 *es_w1t[EXP], *es_w2t[EXP], *es_w3t[EXP];
  for (int e = 0; e < EXP; ++e) {
    et_w1t[e] = allocH(256 * 128);
    et_w2t[e] = allocH(256 * 256);
    et_w3t[e] = allocH(128 * 256);
    es_w1t[e] = allocH(128 * 128);
    es_w2t[e] = allocH(64 * 128);
    es_w3t[e] = allocH(16 * 64);
  }
  float* ws_sh_traj  = allocF(N * TC);
  float* ws_sh_score = allocF(N);
  float* ws_tkp      = allocF(N * 2);
  int*   ws_tki      = (int*)allocB(N * 2 * sizeof(int));
  const int NBLK = NTOK / 256;
  float* ws_partials = allocF((size_t)NBLK * EXP);

  auto conv = [&](const float* src, _Float16* dst, int K, int Nn, int Np) {
    int total = Np * K;
    convert_w<<<dim3((total + 255) / 256), dim3(256), 0, stream>>>(src, dst, K, Nn, Np);
  };

  // router / shared weights
  conv(rw1, r_w1t, 128, 256, 256);  conv(rw2, r_w2t, 256, 128, 128);  conv(rw3, r_w3t, 128, 5, 16);
  conv(stw1, st_w1t, 128, 256, 256); conv(stw2, st_w2t, 256, 256, 256); conv(stw3, st_w3t, 256, 120, 128);
  conv(ssw1, ss_w1t, 128, 128, 128); conv(ssw2, ss_w2t, 128, 64, 64);   conv(ssw3, ss_w3t, 64, 1, 16);
  for (int e = 0; e < EXP; ++e) {
    conv(etw1 + (size_t)e * 128 * 256, et_w1t[e], 128, 256, 256);
    conv(etw2 + (size_t)e * 256 * 256, et_w2t[e], 256, 256, 256);
    conv(etw3 + (size_t)e * 256 * 120, et_w3t[e], 256, 120, 128);
    conv(esw1 + (size_t)e * 128 * 128, es_w1t[e], 128, 128, 128);
    conv(esw2 + (size_t)e * 128 * 64,  es_w2t[e], 128, 64, 64);
    conv(esw3 + (size_t)e * 64,        es_w3t[e], 64, 1, 16);
  }

  auto mlp = [&](const _Float16* w1t, const float* b1, int H1,
                 const _Float16* w2t, const float* b2, int H2,
                 const _Float16* w3t, const float* b3, int H3p, int H3r,
                 float* o, int ostride) {
    mlp3_wmma<<<dim3(NTOK / ROWS), dim3(NTHR), 0, stream>>>(
        x, w1t, b1, H1, w2t, b2, H2, w3t, b3, H3p, H3r, o, ostride);
  };

  // router logits directly into output
  mlp(r_w1t, rb1, 256, r_w2t, rb2, 128, r_w3t, rb3, 16, EXP, o_logits, EXP);
  // shared expert
  mlp(st_w1t, stb1, 256, st_w2t, stb2, 256, st_w3t, stb3, 128, TC, ws_sh_traj, TC);
  mlp(ss_w1t, ssb1, 128, ss_w2t, ssb2, 64, ss_w3t, ssb3, 16, 1, ws_sh_score, 1);
  // dense expert stacks, directly into traj_all / score_all outputs
  for (int e = 0; e < EXP; ++e) {
    mlp(et_w1t[e], etb1 + (size_t)e * 256, 256,
        et_w2t[e], etb2 + (size_t)e * 256, 256,
        et_w3t[e], etb3 + (size_t)e * TC, 128, TC,
        o_traj_all + (size_t)e * N * TC, TC);
    mlp(es_w1t[e], esb1 + (size_t)e * 128, 128,
        es_w2t[e], esb2 + (size_t)e * 64, 64,
        es_w3t[e], esb3 + (size_t)e, 16, 1,
        o_score_all + (size_t)e * N, 1);
  }

  router_post<<<dim3(NBLK), dim3(256), 0, stream>>>(o_logits, ws_tkp, ws_tki, o_topk, ws_partials);
  aux_kernel<<<dim3(1), dim3(32), 0, stream>>>(ws_partials, NBLK, o_aux);
  combine_kernel<<<dim3(NTOK / 8), dim3(256), 0, stream>>>(
      ws_tki, ws_tkp, ws_sh_traj, ws_sh_score, o_traj_all, o_score_all, o_ftraj, o_fscore);
}
